// EquivariantUpdate_987842478183
// MI455X (gfx1250) — matrix operand order repaired and checked by reading, not compiled, not run on hardware
//
#include <hip/hip_runtime.h>

#define HIDN 128
#define NEDGE 800000
#define NNODE 50000
#define WAVES 4
#define XSTRIDE 136   // halves per row in per-wave x tile (128 + pad, keeps 16B row alignment)

typedef _Float16 v16h __attribute__((ext_vector_type(16)));
typedef _Float16 v8h  __attribute__((ext_vector_type(8)));
typedef float    v8f  __attribute__((ext_vector_type(8)));

#define WMMA_F16(a, b, c) \
  __builtin_amdgcn_wmma_f32_16x16x32_f16(false, (a), false, (b), (short)0, (c), false, false)

static __device__ __forceinline__ v16h cat16(v8h lo, v8h hi) {
  return __builtin_shufflevector(lo, hi, 0,1,2,3,4,5,6,7,8,9,10,11,12,13,14,15);
}

// B fragment (32x16 f16): lane holds column n=lane%16; halves are K = hh*8 .. hh*8+7
// and 16+hh*8 .. 16+hh*8+7 in the transposed-weight row (matches A layout).
static __device__ __forceinline__ v16h loadB(const _Float16* rowp, int hh) {
  const _Float16* p = rowp + hh * 8;
  return cat16(*(const v8h*)p, *(const v8h*)(p + 16));
}

// A fragment (16x32 f16) from LDS: caller passes rowptr + k0 + hh*8;
// halves are [p..p+7] and [p+16..p+23] per the ISA A layout.
static __device__ __forceinline__ v16h loadA(const _Float16* p) {
  return cat16(*(const v8h*)p, *(const v8h*)(p + 16));
}

// A fragment built from global f32 (gathered h row), same K grouping, cvt f32->f16.
static __device__ __forceinline__ v16h buildA_f32(const float* src) {
  float4 a0 = *(const float4*)(src);
  float4 a1 = *(const float4*)(src + 4);
  float4 a2 = *(const float4*)(src + 16);
  float4 a3 = *(const float4*)(src + 20);
  v16h r;
  r[0]=(_Float16)a0.x;  r[1]=(_Float16)a0.y;  r[2]=(_Float16)a0.z;  r[3]=(_Float16)a0.w;
  r[4]=(_Float16)a1.x;  r[5]=(_Float16)a1.y;  r[6]=(_Float16)a1.z;  r[7]=(_Float16)a1.w;
  r[8]=(_Float16)a2.x;  r[9]=(_Float16)a2.y;  r[10]=(_Float16)a2.z; r[11]=(_Float16)a2.w;
  r[12]=(_Float16)a3.x; r[13]=(_Float16)a3.y; r[14]=(_Float16)a3.z; r[15]=(_Float16)a3.w;
  return r;
}

// ---- kernel 0: zero the output accumulator --------------------------------
__global__ void eq_zero_kernel(float* __restrict__ out, int n) {
  int i = blockIdx.x * blockDim.x + threadIdx.x;
  if (i < n) out[i] = 0.f;
}

// ---- kernel 1: convert + transpose weights to f16 in workspace ------------
// ws layout (halves): W1t[128][256] | W2t[128][128] | W3t[16][128]
__global__ void eq_prep_kernel(const float* __restrict__ W1, const float* __restrict__ W2,
                               const float* __restrict__ W3, _Float16* __restrict__ ws) {
  int i = blockIdx.x * blockDim.x + threadIdx.x;
  if (i < 32768) {                       // W1t[n][k] = W1[k][n], k<256
    int n = i >> 8, k = i & 255;
    ws[i] = (_Float16)W1[k * 128 + n];
  } else if (i < 49152) {                // W2t[n][k] = W2[k][n]
    int j = i - 32768; int n = j >> 7, k = j & 127;
    ws[i] = (_Float16)W2[k * 128 + n];
  } else if (i < 51200) {                // W3t[p][k] = W3[k][p] (p<9), zero pad p=9..15
    int j = i - 49152; int p = j >> 7, k = j & 127;
    ws[i] = (_Float16)((p < 9) ? W3[k * 9 + p] : 0.f);
  }
}

// ---- kernel 2: fused edge MLP (WMMA) + LN + SiLU + phi^T*cd + atomic scatter
__global__ __launch_bounds__(128) void eq_edge_kernel(
    const float* __restrict__ h, const float* __restrict__ edge_attr,
    const float* __restrict__ edge_mask, const int* __restrict__ row,
    const int* __restrict__ col, const float* __restrict__ coord_diff,
    const float* __restrict__ W1full, const float* __restrict__ b1,
    const float* __restrict__ g1, const float* __restrict__ be1,
    const float* __restrict__ b2, const float* __restrict__ g2,
    const float* __restrict__ be2, const _Float16* __restrict__ wsw,
    float* __restrict__ out)
{
  extern __shared__ unsigned char smem[];
  _Float16* sW1t = (_Float16*)smem;          // [128][256]
  _Float16* sW2t = sW1t + 128 * 256;         // [128][128]
  _Float16* sW3t = sW2t + 128 * 128;         // [16][128]
  float* sv = (float*)(smem + (size_t)(128*256 + 128*128 + 16*128) * 2);
  float* s_w1l = sv;        float* s_b1 = sv + 128; float* s_g1 = sv + 256;
  float* s_be1 = sv + 384;  float* s_b2 = sv + 512; float* s_g2 = sv + 640;
  float* s_be2 = sv + 768;
  _Float16* xbase = (_Float16*)(sv + 896);           // WAVES * [16][XSTRIDE] f16
  float* phibase = (float*)(xbase + WAVES * 16 * XSTRIDE);  // WAVES * [16][12] f32

  const int tid = threadIdx.x;
  const int wave = tid >> 5, lane = tid & 31;
  const int lm = lane & 15, hh = lane >> 4;

  // Stage f16 weights (bulk 16B copies) + f32 vectors into LDS.
  {
    const uint4* gs = (const uint4*)wsw;
    uint4* ls = (uint4*)smem;
    const int nvec = (128*256 + 128*128 + 16*128) * 2 / 16;  // 6400
    for (int i = tid; i < nvec; i += 128) ls[i] = gs[i];
    for (int i = tid; i < 896; i += 128) {
      int wv = i >> 7, j = i & 127; float v;
      switch (wv) {
        case 0: v = W1full[256 * 128 + j]; break;   // W1 last input row (edge_attr column weights)
        case 1: v = b1[j]; break;
        case 2: v = g1[j]; break;
        case 3: v = be1[j]; break;
        case 4: v = b2[j]; break;
        case 5: v = g2[j]; break;
        default: v = be2[j]; break;
      }
      sv[i] = v;
    }
  }
  __syncthreads();

  _Float16* xbuf = xbase + wave * 16 * XSTRIDE;
  float* phibuf = phibase + wave * 16 * 12;

  const int e0 = (blockIdx.x * WAVES + wave) * 16;
  const int eL = min(e0 + lm, NEDGE - 1);
  const int r = row[eL], c = col[eL];
  const float* hr = h + (size_t)r * HIDN;
  const float* hc = h + (size_t)c * HIDN;

  // Prefetch the post-MLP scatter operands (consumed ~100 WMMAs from now) so
  // they are cache-resident by then: emits global_prefetch_b8 on gfx1250.
  __builtin_prefetch(coord_diff + (size_t)eL * 9, 0, 3);
  __builtin_prefetch(edge_mask + eL, 0, 3);

  float eav[8];
  #pragma unroll
  for (int v = 0; v < 8; ++v) eav[v] = edge_attr[min(e0 + v + hh * 8, NEDGE - 1)];

  const v8f vzero = {0.f,0.f,0.f,0.f,0.f,0.f,0.f,0.f};
  v8f acc[8];
  #pragma unroll
  for (int t = 0; t < 8; ++t) acc[t] = vzero;

  // ---------------- Layer 1: [16,256] @ [256,128] (h[row] || h[col]) -------
  #pragma unroll
  for (int ks = 0; ks < 8; ++ks) {
    const float* src = (ks < 4) ? (hr + ks * 32 + hh * 8) : (hc + (ks - 4) * 32 + hh * 8);
    v16h a = buildA_f32(src);
    #pragma unroll
    for (int t = 0; t < 8; ++t) {
      v16h b = loadB(sW1t + (t * 16 + lm) * 256 + ks * 32, hh);
      acc[t] = WMMA_F16(a, b, acc[t]);
    }
  }
  // bias + edge_attr rank-1 term; then LayerNorm + SiLU in C-tile layout
  #pragma unroll
  for (int t = 0; t < 8; ++t) {
    const int n = t * 16 + lm;
    const float bias = s_b1[n], wl = s_w1l[n];
    #pragma unroll
    for (int v = 0; v < 8; ++v) acc[t][v] += bias + eav[v] * wl;
  }
  {
    float mu[8], ri[8];
    #pragma unroll
    for (int v = 0; v < 8; ++v) {
      float p = 0.f, q = 0.f;
      #pragma unroll
      for (int t = 0; t < 8; ++t) { float x = acc[t][v]; p += x; q += x * x; }
      #pragma unroll
      for (int m = 8; m >= 1; m >>= 1) { p += __shfl_xor(p, m, 32); q += __shfl_xor(q, m, 32); }
      float mean = p * (1.f / HIDN);
      mu[v] = mean;
      ri[v] = rsqrtf(q * (1.f / HIDN) - mean * mean + 1e-5f);
    }
    #pragma unroll
    for (int t = 0; t < 8; ++t) {
      const int n = t * 16 + lm;
      const float gg = s_g1[n], bb = s_be1[n];
      #pragma unroll
      for (int v = 0; v < 8; ++v) {
        float x = (acc[t][v] - mu[v]) * ri[v] * gg + bb;
        x = x / (1.f + __expf(-x));               // SiLU
        xbuf[(v + hh * 8) * XSTRIDE + n] = (_Float16)x;
      }
    }
  }
  __syncthreads();

  // ---------------- Layer 2: [16,128] @ [128,128] --------------------------
  #pragma unroll
  for (int t = 0; t < 8; ++t) acc[t] = vzero;
  const _Float16* xrow = xbuf + lm * XSTRIDE;
  #pragma unroll
  for (int ks = 0; ks < 4; ++ks) {
    v16h a = loadA(xrow + ks * 32 + hh * 8);
    #pragma unroll
    for (int t = 0; t < 8; ++t) {
      v16h b = loadB(sW2t + (t * 16 + lm) * 128 + ks * 32, hh);
      acc[t] = WMMA_F16(a, b, acc[t]);
    }
  }
  {
    float mu[8], ri[8];
    #pragma unroll
    for (int t = 0; t < 8; ++t) {
      const float bias = s_b2[t * 16 + lm];
      #pragma unroll
      for (int v = 0; v < 8; ++v) acc[t][v] += bias;
    }
    #pragma unroll
    for (int v = 0; v < 8; ++v) {
      float p = 0.f, q = 0.f;
      #pragma unroll
      for (int t = 0; t < 8; ++t) { float x = acc[t][v]; p += x; q += x * x; }
      #pragma unroll
      for (int m = 8; m >= 1; m >>= 1) { p += __shfl_xor(p, m, 32); q += __shfl_xor(q, m, 32); }
      float mean = p * (1.f / HIDN);
      mu[v] = mean;
      ri[v] = rsqrtf(q * (1.f / HIDN) - mean * mean + 1e-5f);
    }
    #pragma unroll
    for (int t = 0; t < 8; ++t) {
      const int n = t * 16 + lm;
      const float gg = s_g2[n], bb = s_be2[n];
      #pragma unroll
      for (int v = 0; v < 8; ++v) {
        float x = (acc[t][v] - mu[v]) * ri[v] * gg + bb;
        x = x / (1.f + __expf(-x));
        xbuf[(v + hh * 8) * XSTRIDE + n] = (_Float16)x;
      }
    }
  }
  __syncthreads();

  // ---------------- Layer 3: [16,128] @ [128,16(pad of 9)] -----------------
  v8f acc3 = vzero;
  #pragma unroll
  for (int ks = 0; ks < 4; ++ks) {
    v16h a = loadA(xrow + ks * 32 + hh * 8);
    v16h b = loadB(sW3t + lm * 128 + ks * 32, hh);
    acc3 = WMMA_F16(a, b, acc3);
  }
  if (lm < 9) {
    #pragma unroll
    for (int v = 0; v < 8; ++v) phibuf[(v + hh * 8) * 12 + lm] = acc3[v];
  }
  __syncthreads();

  // ---------------- trans = phi^T * coord_diff, masked, atomic segment-sum -
  // Both 16-lane halves work on the same edge (lane % 16); the 9 outputs are
  // split even/odd across halves so all 32 lanes issue atomics.
  {
    const int e2 = e0 + lm;
    if (e2 < NEDGE) {
      const float em = edge_mask[e2];
      const float* cd = coord_diff + (size_t)e2 * 9;
      float cdv[9], ph[9];
      #pragma unroll
      for (int j = 0; j < 9; ++j) cdv[j] = cd[j];
      #pragma unroll
      for (int j = 0; j < 9; ++j) ph[j] = phibuf[lm * 12 + j];
      float* dst = out + (size_t)r * 9;
      #pragma unroll
      for (int idx = 0; idx < 9; ++idx) {
        if ((idx & 1) == hh) {
          const int l = idx / 3, k = idx - l * 3;
          float t = cdv[k] * ph[l] + cdv[3 + k] * ph[3 + l] + cdv[6 + k] * ph[6 + l];
          atomicAdd(dst + idx, t * em);
        }
      }
    }
  }
}

// ---- kernel 3: out = (coord + agg/100) * node_mask ------------------------
__global__ void eq_final_kernel(const float* __restrict__ coord,
                                const float* __restrict__ node_mask,
                                float* __restrict__ out, int n) {
  int i = blockIdx.x * blockDim.x + threadIdx.x;
  if (i < n) out[i] = (coord[i] + out[i] * 0.01f) * node_mask[i / 9];
}

extern "C" void kernel_launch(void* const* d_in, const int* in_sizes, int n_in,
                              void* d_out, int out_size, void* d_ws, size_t ws_size,
                              hipStream_t stream) {
  const float* h          = (const float*)d_in[0];
  const float* coord      = (const float*)d_in[1];
  const float* coord_diff = (const float*)d_in[2];
  const float* edge_attr  = (const float*)d_in[3];
  const float* edge_mask  = (const float*)d_in[4];
  const float* node_mask  = (const float*)d_in[5];
  const int*   row        = (const int*)d_in[6];
  const int*   col        = (const int*)d_in[7];
  const float* W1  = (const float*)d_in[8];
  const float* b1  = (const float*)d_in[9];
  const float* g1  = (const float*)d_in[10];
  const float* be1 = (const float*)d_in[11];
  const float* W2  = (const float*)d_in[12];
  const float* b2  = (const float*)d_in[13];
  const float* g2  = (const float*)d_in[14];
  const float* be2 = (const float*)d_in[15];
  const float* W3  = (const float*)d_in[16];
  float* out = (float*)d_out;
  _Float16* wsw = (_Float16*)d_ws;

  const int on = NNODE * 9;
  eq_zero_kernel<<<(on + 255) / 256, 256, 0, stream>>>(out, on);
  eq_prep_kernel<<<200, 256, 0, stream>>>(W1, W2, W3, wsw);

  const size_t smem = (size_t)(128 * 256 + 128 * 128 + 16 * 128) * 2  // f16 weights
                    + 896 * 4                                          // f32 vectors
                    + (size_t)WAVES * 16 * XSTRIDE * 2                 // x tiles
                    + (size_t)WAVES * 16 * 12 * 4;                     // phi tiles
  const int tiles = (NEDGE + 15) / 16;
  const int blocks = (tiles + WAVES - 1) / WAVES;
  eq_edge_kernel<<<blocks, 128, smem, stream>>>(h, edge_attr, edge_mask, row, col,
                                                coord_diff, W1, b1, g1, be1, b2, g2,
                                                be2, wsw, out);

  eq_final_kernel<<<(on + 255) / 256, 256, 0, stream>>>(coord, node_mask, out, on);
}
